// GraphConvolution_88502096101454
// MI455X (gfx1250) — matrix operand order repaired and checked by reading, not compiled, not run on hardware
//
#include <hip/hip_runtime.h>
#include <hip/hip_bf16.h>
#include <stdint.h>

#define N_NODES 100000
#define E_EDGES 3200000
#define IN_F    256
#define OUT_F   256
#define CLAMP_V 10.0f
#define EDGES_PER_WAVE 256

typedef __attribute__((ext_vector_type(16))) __bf16 v16bf;
typedef __attribute__((ext_vector_type(8)))  float  v8f;

__device__ __forceinline__ float clampf(float v) {
    return fminf(fmaxf(v, -CLAMP_V), CLAMP_V);
}

// float -> bf16 bits, round-to-nearest-even
__device__ __forceinline__ uint16_t f2bfbits(float f) {
    union { float f; uint32_t u; } v; v.f = f;
    uint32_t r = v.u + 0x7FFFu + ((v.u >> 16) & 1u);
    return (uint16_t)(r >> 16);
}

__device__ __forceinline__ __bf16 f2bf(float f) {
    uint16_t h = f2bfbits(f);
    __bf16 o; __builtin_memcpy(&o, &h, 2); return o;
}

// ---------------------------------------------------------------------------
// Kernel 1: W [k][n] fp32 -> Wt [n][k] bf16 (so WMMA B fragments are
// contiguous 32B per lane).
// ---------------------------------------------------------------------------
__global__ __launch_bounds__(256) void prep_weight(const float* __restrict__ w,
                                                   uint16_t* __restrict__ wt) {
    int gid = blockIdx.x * 256 + threadIdx.x;   // 65536 total
    int k = gid & 255;
    int n = gid >> 8;
    wt[n * IN_F + k] = f2bfbits(w[k * OUT_F + n]);
}

// ---------------------------------------------------------------------------
// Kernel 2: support = clip(clip(x) @ W)  via bf16 WMMA, stored as bf16.
// One wave per 16-row tile; 16 column tiles x 8 K-steps = 128 WMMAs/wave.
// ---------------------------------------------------------------------------
__global__ __launch_bounds__(256) void gemm_support(const float* __restrict__ x,
                                                    const __bf16* __restrict__ wt,
                                                    uint16_t* __restrict__ sup) {
    const int lane = threadIdx.x & 31;
    const int wave = threadIdx.x >> 5;
    const int tile = blockIdx.x * 8 + wave;
    if (tile >= N_NODES / 16) return;

    const int r0 = tile * 16;
    const int m  = lane & 15;
    const int hi = lane >> 4;            // 0 for lanes 0-15, 1 for 16-31

    v8f acc[16];
    const v8f vz = {0.f,0.f,0.f,0.f,0.f,0.f,0.f,0.f};
#pragma unroll
    for (int n = 0; n < 16; ++n) acc[n] = vz;

    const float* xrow = x + (size_t)(r0 + m) * IN_F;

#pragma unroll
    for (int ks = 0; ks < 8; ++ks) {
        const int k0 = ks * 32;
        // A fragment: 16x32, contiguous k order (B uses the same order).
        // lanes 0-15 : elems 0-15 = K k0..k0+15
        // lanes 16-31: elems 0-15 = K k0+16..k0+31
        const int ka = k0 + hi * 16;
        float4 f0 = *(const float4*)(xrow + ka);
        float4 f1 = *(const float4*)(xrow + ka + 4);
        float4 f2 = *(const float4*)(xrow + ka + 8);
        float4 f3 = *(const float4*)(xrow + ka + 12);
        v16bf a;
        a[0]  = f2bf(clampf(f0.x)); a[1]  = f2bf(clampf(f0.y));
        a[2]  = f2bf(clampf(f0.z)); a[3]  = f2bf(clampf(f0.w));
        a[4]  = f2bf(clampf(f1.x)); a[5]  = f2bf(clampf(f1.y));
        a[6]  = f2bf(clampf(f1.z)); a[7]  = f2bf(clampf(f1.w));
        a[8]  = f2bf(clampf(f2.x)); a[9]  = f2bf(clampf(f2.y));
        a[10] = f2bf(clampf(f2.z)); a[11] = f2bf(clampf(f2.w));
        a[12] = f2bf(clampf(f3.x)); a[13] = f2bf(clampf(f3.y));
        a[14] = f2bf(clampf(f3.z)); a[15] = f2bf(clampf(f3.w));

        // B fragment: lanes 0-15 hold K k0..k0+15 at N=lane;
        // lanes 16-31 hold K k0+16..k0+31 at N=lane-16.  Wt is [n][k] bf16,
        // so this is one contiguous 32B load.
        const int kb = k0 + hi * 16;
#pragma unroll
        for (int n = 0; n < 16; ++n) {
            v16bf b = *(const v16bf*)(wt + (size_t)(n * 16 + m) * IN_F + kb);
            acc[n] = __builtin_amdgcn_wmma_f32_16x16x32_bf16(
                false, a, false, b, (short)0, acc[n], false, false);
        }
    }

    // C/D layout: VGPR i holds row (i + 8*hi), col = n*16 + (lane&15).
#pragma unroll
    for (int n = 0; n < 16; ++n) {
#pragma unroll
        for (int i = 0; i < 8; ++i) {
            int row = r0 + i + hi * 8;
            int col = n * 16 + m;
            sup[(size_t)row * OUT_F + col] = f2bfbits(clampf(acc[n][i]));
        }
    }
}

// ---------------------------------------------------------------------------
// Kernel 3: segmented SpMM.  One wave walks EDGES_PER_WAVE sorted edges;
// each lane owns 8 columns (32 lanes x 8 = 256).  bf16 gathers (L2-resident),
// fp32 register accumulation, atomic flush at row boundaries.
// ---------------------------------------------------------------------------
__global__ __launch_bounds__(256) void spmm_kernel(const int* __restrict__ rows,
                                                   const int* __restrict__ cols,
                                                   const float* __restrict__ vals,
                                                   const uint32_t* __restrict__ sup32,
                                                   float* __restrict__ out) {
    const int lane = threadIdx.x & 31;
    const long wv  = (long)((blockIdx.x * blockDim.x + threadIdx.x) >> 5);
    long e0 = wv * EDGES_PER_WAVE;
    if (e0 >= E_EDGES) return;
    long e1 = e0 + EDGES_PER_WAVE;
    if (e1 > E_EDGES) e1 = E_EDGES;

    float acc[8];
#pragma unroll
    for (int j = 0; j < 8; ++j) acc[j] = 0.f;
    int cur = -1;

    for (long e = e0; e < e1; ++e) {
        int r = rows[e];
        if (r != cur) {
            if (cur >= 0) {
                float* o = out + (size_t)cur * OUT_F + lane * 8;
#pragma unroll
                for (int j = 0; j < 8; ++j) atomicAdd(o + j, acc[j]);
            }
            cur = r;
#pragma unroll
            for (int j = 0; j < 8; ++j) acc[j] = 0.f;
        }
        int c = cols[e];
        float v = vals[e];
        const uint4 pk = *(const uint4*)(sup32 + (size_t)c * (OUT_F / 2) + lane * 4);
        acc[0] = fmaf(v, __uint_as_float(pk.x << 16),          acc[0]);
        acc[1] = fmaf(v, __uint_as_float(pk.x & 0xFFFF0000u),  acc[1]);
        acc[2] = fmaf(v, __uint_as_float(pk.y << 16),          acc[2]);
        acc[3] = fmaf(v, __uint_as_float(pk.y & 0xFFFF0000u),  acc[3]);
        acc[4] = fmaf(v, __uint_as_float(pk.z << 16),          acc[4]);
        acc[5] = fmaf(v, __uint_as_float(pk.z & 0xFFFF0000u),  acc[5]);
        acc[6] = fmaf(v, __uint_as_float(pk.w << 16),          acc[6]);
        acc[7] = fmaf(v, __uint_as_float(pk.w & 0xFFFF0000u),  acc[7]);
    }
    if (cur >= 0) {
        float* o = out + (size_t)cur * OUT_F + lane * 8;
#pragma unroll
        for (int j = 0; j < 8; ++j) atomicAdd(o + j, acc[j]);
    }
}

// ---------------------------------------------------------------------------
// Kernel 4: out = clip(relu(clip(out) + bias))
// ---------------------------------------------------------------------------
__global__ __launch_bounds__(256) void finalize_kernel(float* __restrict__ out,
                                                       const float* __restrict__ bias) {
    size_t i = (size_t)blockIdx.x * 256 + threadIdx.x;   // over N*64 float4s
    if (i >= (size_t)N_NODES * (OUT_F / 4)) return;
    float4* o4 = (float4*)out;
    float4 v = o4[i];
    int cb = ((int)(i & 63)) * 4;
    v.x = clampf(fmaxf(clampf(v.x) + bias[cb + 0], 0.f));
    v.y = clampf(fmaxf(clampf(v.y) + bias[cb + 1], 0.f));
    v.z = clampf(fmaxf(clampf(v.z) + bias[cb + 2], 0.f));
    v.w = clampf(fmaxf(clampf(v.w) + bias[cb + 3], 0.f));
    o4[i] = v;
}

// ---------------------------------------------------------------------------
extern "C" void kernel_launch(void* const* d_in, const int* in_sizes, int n_in,
                              void* d_out, int out_size, void* d_ws, size_t ws_size,
                              hipStream_t stream) {
    (void)in_sizes; (void)n_in; (void)ws_size;
    const float* x    = (const float*)d_in[0];
    const int*   rows = (const int*)d_in[1];
    const int*   cols = (const int*)d_in[2];
    const float* vals = (const float*)d_in[3];
    const float* w    = (const float*)d_in[4];
    const float* bias = (const float*)d_in[5];
    float* out = (float*)d_out;

    // workspace: [0, 51.2MB) support bf16, then 128KB transposed bf16 weight
    uint16_t* sup = (uint16_t*)d_ws;
    const size_t sup_bytes = (size_t)N_NODES * OUT_F * sizeof(uint16_t);
    uint16_t* wt = (uint16_t*)((char*)d_ws + sup_bytes);

    hipMemsetAsync(d_out, 0, (size_t)out_size * sizeof(float), stream);

    prep_weight<<<(IN_F * OUT_F) / 256, 256, 0, stream>>>(w, wt);

    const int tiles = N_NODES / 16;                 // 6250
    const int gemm_blocks = (tiles + 7) / 8;        // 8 waves per block
    gemm_support<<<gemm_blocks, 256, 0, stream>>>(x, (const __bf16*)wt, sup);

    const long waves = (E_EDGES + EDGES_PER_WAVE - 1) / EDGES_PER_WAVE;
    const int spmm_blocks = (int)((waves + 7) / 8);
    spmm_kernel<<<spmm_blocks, 256, 0, stream>>>(rows, cols, vals,
                                                 (const uint32_t*)sup, out);

    const size_t fin_threads = (size_t)N_NODES * (OUT_F / 4);
    const int fin_blocks = (int)((fin_threads + 255) / 256);
    finalize_kernel<<<fin_blocks, 256, 0, stream>>>(out, bias);
}
